// Block_6554120094246
// MI455X (gfx1250) — matrix operand order repaired and checked by reading, not compiled, not run on hardware
//
#include <hip/hip_runtime.h>
#include <hip/hip_bf16.h>
#include <stdint.h>

// ---------------------------------------------------------------------------
// Types for CDNA5 WMMA
// ---------------------------------------------------------------------------
typedef __attribute__((ext_vector_type(16))) __bf16 v16bf;
typedef __attribute__((ext_vector_type(8)))  float  v8f;

union FragBF { v16bf v; uint4 q[2]; unsigned short h[16]; };

__device__ __forceinline__ v8f vzero8() {
  v8f z = {0.f,0.f,0.f,0.f,0.f,0.f,0.f,0.f};
  return z;
}

#define C_DIM 768
#define NH    12
#define HD    64
#define ATT_SCALE 0.125f   // 64^-0.5
#define LN_EPS 1e-5f

__device__ __forceinline__ unsigned short f2bf(float f) {
  union { float f; uint32_t u; } c; c.f = f;
  uint32_t u = c.u;
  u += 0x7FFFu + ((u >> 16) & 1u);      // round-to-nearest-even
  return (unsigned short)(u >> 16);
}
__device__ __forceinline__ float bf2f(unsigned short h) {
  union { uint32_t u; float f; } c; c.u = ((uint32_t)h) << 16; return c.f;
}

// ---------------------------------------------------------------------------
// CDNA5 async global->LDS copy (ASYNCcnt-tracked, no VGPR round trip).
// LDS aperture keeps the byte offset in addr[31:0], so truncating the generic
// pointer yields the LDS address the instruction wants in VDST.
// ---------------------------------------------------------------------------
__device__ __forceinline__ void async_cp16(void* lds_ptr, const void* gptr) {
  unsigned loff = (unsigned)(uintptr_t)lds_ptr;
  asm volatile("global_load_async_to_lds_b128 %0, %1, off"
               :: "v"(loff), "v"(gptr) : "memory");
}
__device__ __forceinline__ void wait_async0() {
  asm volatile("s_wait_asynccnt 0" ::: "memory");
}

// ---------------------------------------------------------------------------
// Elementwise helpers
// ---------------------------------------------------------------------------
__global__ void k_f32_to_bf16(const float* __restrict__ x, unsigned short* __restrict__ y, int n) {
  int i = blockIdx.x * blockDim.x + threadIdx.x;
  int stride = gridDim.x * blockDim.x;
  for (; i < n; i += stride) y[i] = f2bf(x[i]);
}

__global__ void k_copy_f32(const float* __restrict__ x, float* __restrict__ y, int n) {
  int i = blockIdx.x * blockDim.x + threadIdx.x;
  int stride = gridDim.x * blockDim.x;
  for (; i < n; i += stride) y[i] = x[i];
}

// dst[0..n0) = p0 (or 0), dst[n0..n0+n1) = p1 (or 0), dst[..+n2) = p2 (or 0)
__global__ void k_bias3(float* __restrict__ dst,
                        const float* p0, int n0,
                        const float* p1, int n1,
                        const float* p2, int n2) {
  int n = n0 + n1 + n2;
  int i = blockIdx.x * blockDim.x + threadIdx.x;
  int stride = gridDim.x * blockDim.x;
  for (; i < n; i += stride) {
    float v = 0.f;
    if (i < n0)            v = p0 ? p0[i] : 0.f;
    else if (i < n0 + n1)  v = p1 ? p1[i - n0] : 0.f;
    else                   v = p2 ? p2[i - n0 - n1] : 0.f;
    dst[i] = v;
  }
}

// ---------------------------------------------------------------------------
// LayerNorm over C=768, fp32 in -> bf16 out. One 256-thread block per row.
// ---------------------------------------------------------------------------
__global__ void k_layernorm_bf16(const float* __restrict__ x,
                                 const float* __restrict__ g,
                                 const float* __restrict__ b,
                                 unsigned short* __restrict__ y, int M) {
  int row = blockIdx.x;
  if (row >= M) return;
  const float* xr = x + (size_t)row * C_DIM;
  int t = threadIdx.x;
  float v0 = xr[t], v1 = xr[t + 256], v2 = xr[t + 512];
  float s  = v0 + v1 + v2;
  float s2 = v0 * v0 + v1 * v1 + v2 * v2;
  for (int o = 16; o > 0; o >>= 1) {
    s  += __shfl_xor(s,  o, 32);
    s2 += __shfl_xor(s2, o, 32);
  }
  __shared__ float rs[8], rs2[8];
  int w = t >> 5, ln = t & 31;
  if (ln == 0) { rs[w] = s; rs2[w] = s2; }
  __syncthreads();
  float ts = 0.f, ts2 = 0.f;
  #pragma unroll
  for (int i = 0; i < 8; i++) { ts += rs[i]; ts2 += rs2[i]; }
  float mean = ts * (1.0f / 768.0f);
  float var  = ts2 * (1.0f / 768.0f) - mean * mean;
  float inv  = rsqrtf(var + LN_EPS);
  unsigned short* yr = y + (size_t)row * C_DIM;
  yr[t]       = f2bf((v0 - mean) * inv * g[t]       + b[t]);
  yr[t + 256] = f2bf((v1 - mean) * inv * g[t + 256] + b[t + 256]);
  yr[t + 512] = f2bf((v2 - mean) * inv * g[t + 512] + b[t + 512]);
}

// ---------------------------------------------------------------------------
// WMMA GEMM: Y[M,N] = X[M,K] @ W[N,K]^T + bias, bf16 inputs, f32 accumulate.
// Block tile 128(M) x 64(N), 8 waves, each wave 32x32 (2x2 WMMA 16x16x32).
// Double-buffered LDS fed by global_load_async_to_lds_b128.
// Epilogues: 0 = bf16 out, 1 = f32 residual add (in place), 2 = QuickGELU
//            (bf16 out), 3 = exact GELU (bf16 out).
// ---------------------------------------------------------------------------
#define EPI_BF16  0
#define EPI_RES   1
#define EPI_QGELU 2
#define EPI_GELU  3

template <int EPI>
__global__ void k_gemm_wmma(const unsigned short* __restrict__ A,
                            const unsigned short* __restrict__ W,
                            const float* __restrict__ bias,
                            unsigned short* __restrict__ Ybf,
                            float* __restrict__ Yres,
                            int M, int N, int K) {
  __shared__ __align__(16) unsigned short As[2][128][40];
  __shared__ __align__(16) unsigned short Bs[2][64][40];

  int tid  = threadIdx.x;
  int wave = tid >> 5, lane = tid & 31;
  int half = lane >> 4, l16 = lane & 15;
  int m0 = blockIdx.y * 128;
  int n0 = blockIdx.x * 64;
  int wm = (wave >> 1) * 32;   // wave M offset in tile
  int wn = (wave & 1) * 32;    // wave N offset in tile

  // per-thread tile-staging coordinates
  int ar0 = tid >> 2,         ac0 = (tid & 3) * 8;          // A chunk 0
  int ar1 = (tid + 256) >> 2, ac1 = (tid & 3) * 8;          // A chunk 1
  int br  = tid >> 2,         bc  = (tid & 3) * 8;          // B chunk

  v8f acc[2][2];
  #pragma unroll
  for (int mi = 0; mi < 2; mi++)
    #pragma unroll
    for (int ni = 0; ni < 2; ni++) acc[mi][ni] = vzero8();

  int kTiles = K >> 5;

  auto stage_tile = [&](int buf, int kb) {
    int gr0 = m0 + ar0; gr0 = gr0 < M ? gr0 : M - 1;
    int gr1 = m0 + ar1; gr1 = gr1 < M ? gr1 : M - 1;
    async_cp16(&As[buf][ar0][ac0], A + (size_t)gr0 * K + kb + ac0);
    async_cp16(&As[buf][ar1][ac1], A + (size_t)gr1 * K + kb + ac1);
    async_cp16(&Bs[buf][br][bc],   W + (size_t)(n0 + br) * K + kb + bc);
  };

  stage_tile(0, 0);
  wait_async0();
  __syncthreads();

  for (int kt = 0; kt < kTiles; ++kt) {
    int cur = kt & 1;
    if (kt + 1 < kTiles) stage_tile(1 - cur, (kt + 1) << 5);

    FragBF a[2], bfr[2];
    #pragma unroll
    for (int mi = 0; mi < 2; mi++) {
      int r = wm + mi * 16 + l16;
      a[mi].q[0] = *reinterpret_cast<const uint4*>(&As[cur][r][half * 8]);
      a[mi].q[1] = *reinterpret_cast<const uint4*>(&As[cur][r][16 + half * 8]);
    }
    #pragma unroll
    for (int ni = 0; ni < 2; ni++) {
      int r = wn + ni * 16 + l16;
      bfr[ni].q[0] = *reinterpret_cast<const uint4*>(&Bs[cur][r][half * 8]);
      bfr[ni].q[1] = *reinterpret_cast<const uint4*>(&Bs[cur][r][16 + half * 8]);
    }
    #pragma unroll
    for (int mi = 0; mi < 2; mi++)
      #pragma unroll
      for (int ni = 0; ni < 2; ni++)
        acc[mi][ni] = __builtin_amdgcn_wmma_f32_16x16x32_bf16(
            false, a[mi].v, false, bfr[ni].v, (short)0, acc[mi][ni], false, false);

    wait_async0();       // next tile landed (this wave's copies)
    __syncthreads();     // all waves: done reading cur, next tile visible
  }

  // Epilogue. Acc layout: lane holds (row = 8*half + j, col = l16) per VGPR j.
  #pragma unroll
  for (int mi = 0; mi < 2; mi++) {
    #pragma unroll
    for (int ni = 0; ni < 2; ni++) {
      int colBase = n0 + wn + ni * 16 + l16;
      float bv = bias ? bias[colBase] : 0.f;
      #pragma unroll
      for (int j = 0; j < 8; j++) {
        int row = m0 + wm + mi * 16 + half * 8 + j;
        if (row < M) {
          float v = acc[mi][ni][j] + bv;
          size_t idx = (size_t)row * N + colBase;
          if (EPI == EPI_RES) {
            Yres[idx] += v;
          } else {
            if (EPI == EPI_QGELU)     v = v / (1.0f + __expf(-1.702f * v));
            else if (EPI == EPI_GELU) v = 0.5f * v * (1.0f + erff(v * 0.70710678118f));
            Ybf[idx] = f2bf(v);
          }
        }
      }
    }
  }
}

// ---------------------------------------------------------------------------
// Flash attention with WMMA (bf16 operands, f32 softmax/accumulate).
// Grid: (ceil(Nq/128), H, B). Block 256 = 8 waves; each wave owns 16 q rows.
// K tile staged with async global->LDS; V staged transposed by hand.
// ---------------------------------------------------------------------------
__global__ void k_flash_wmma(const unsigned short* __restrict__ Qb, int CQ, int qoff0,
                             const unsigned short* __restrict__ Kb, int CK, int koff0,
                             const unsigned short* __restrict__ Vb, int CV, int voff0,
                             unsigned short* __restrict__ Ob,
                             int Nq, int Nk) {
  __shared__ __align__(16) unsigned short Ks[32][72];      // [token][hd]
  __shared__ __align__(16) unsigned short Vt[64][40];      // [hd][token] (transposed)
  __shared__ __align__(16) unsigned short Ps[8][16][40];   // per-wave P tile

  int b = blockIdx.z, h = blockIdx.y;
  int qoff = qoff0 + h * HD, koff = koff0 + h * HD, voff = voff0 + h * HD;
  int tid  = threadIdx.x;
  int wave = tid >> 5, lane = tid & 31, half = lane >> 4, l16 = lane & 15;

  // Q fragments (A-layout), rows clamped
  int qrowA = blockIdx.x * 128 + wave * 16 + l16;
  int qclamp = qrowA < Nq ? qrowA : Nq - 1;
  const unsigned short* qp = Qb + ((size_t)(b * Nq + qclamp)) * CQ + qoff;
  FragBF qa[2];
  #pragma unroll
  for (int s = 0; s < 2; s++) {
    qa[s].q[0] = *reinterpret_cast<const uint4*>(qp + s * 32 + half * 8);
    qa[s].q[1] = *reinterpret_cast<const uint4*>(qp + s * 32 + 16 + half * 8);
  }

  v8f o[4];
  #pragma unroll
  for (int i = 0; i < 4; i++) o[i] = vzero8();
  float mrow[8], lrow[8];
  #pragma unroll
  for (int j = 0; j < 8; j++) { mrow[j] = -3.0e38f; lrow[j] = 0.f; }

  int tok = tid >> 3;
  int d0  = (tid & 7) * 8;

  int kTiles = (Nk + 31) >> 5;
  for (int kt = 0; kt < kTiles; ++kt) {
    int k0 = kt << 5;
    __syncthreads();   // previous tile fully consumed
    {
      int gk = k0 + tok; int gkc = gk < Nk ? gk : Nk - 1;
      // K tile (32x64): async straight into LDS
      async_cp16(&Ks[tok][d0],
                 Kb + ((size_t)(b * Nk + gkc)) * CK + koff + d0);
      // V tile transposed into LDS (needs per-element shuffle)
      union { uint4 q; unsigned short s[8]; } vd;
      vd.q = *reinterpret_cast<const uint4*>(Vb + ((size_t)(b * Nk + gkc)) * CV + voff + d0);
      #pragma unroll
      for (int i = 0; i < 8; i++) Vt[d0 + i][tok] = vd.s[i];
    }
    wait_async0();
    __syncthreads();

    // S = Q K^T  (gemm-K = HD, 2 steps of 32; gemm-N = 32 tokens, 2 tiles)
    v8f s_acc[2] = { vzero8(), vzero8() };
    #pragma unroll
    for (int s = 0; s < 2; s++) {
      FragBF kb_[2];
      #pragma unroll
      for (int ni = 0; ni < 2; ni++) {
        int r = ni * 16 + l16;   // token within tile
        kb_[ni].q[0] = *reinterpret_cast<const uint4*>(&Ks[r][s * 32 + half * 8]);
        kb_[ni].q[1] = *reinterpret_cast<const uint4*>(&Ks[r][s * 32 + 16 + half * 8]);
      }
      #pragma unroll
      for (int ni = 0; ni < 2; ni++)
        s_acc[ni] = __builtin_amdgcn_wmma_f32_16x16x32_bf16(
            false, qa[s].v, false, kb_[ni].v, (short)0, s_acc[ni], false, false);
    }

    // scale + mask + online softmax (row = 8*half + j, spread over 16 lanes)
    float sv[2][8];
    #pragma unroll
    for (int ni = 0; ni < 2; ni++) {
      int col = k0 + ni * 16 + l16;
      bool oob = col >= Nk;
      #pragma unroll
      for (int j = 0; j < 8; j++)
        sv[ni][j] = oob ? -3.0e38f : s_acc[ni][j] * ATT_SCALE;
    }
    #pragma unroll
    for (int j = 0; j < 8; j++) {
      float mx = fmaxf(sv[0][j], sv[1][j]);
      for (int off = 8; off >= 1; off >>= 1) mx = fmaxf(mx, __shfl_xor(mx, off, 32));
      float mnew = fmaxf(mrow[j], mx);
      float corr = __expf(mrow[j] - mnew);
      float p0 = __expf(sv[0][j] - mnew);
      float p1 = __expf(sv[1][j] - mnew);
      float ps = p0 + p1;
      for (int off = 8; off >= 1; off >>= 1) ps += __shfl_xor(ps, off, 32);
      lrow[j] = lrow[j] * corr + ps;
      mrow[j] = mnew;
      #pragma unroll
      for (int ni = 0; ni < 4; ni++) o[ni][j] *= corr;
      Ps[wave][8 * half + j][l16]      = f2bf(p0);
      Ps[wave][8 * half + j][16 + l16] = f2bf(p1);
    }
    asm volatile("s_wait_dscnt 0" ::: "memory");  // LDS cross-lane handoff

    // O += P V  (A = P 16x32 from LDS, B = V^T tiles from Vt)
    FragBF pa;
    pa.q[0] = *reinterpret_cast<const uint4*>(&Ps[wave][l16][half * 8]);
    pa.q[1] = *reinterpret_cast<const uint4*>(&Ps[wave][l16][16 + half * 8]);
    #pragma unroll
    for (int ni = 0; ni < 4; ni++) {
      FragBF vb_;
      int r = ni * 16 + l16;    // hd index
      vb_.q[0] = *reinterpret_cast<const uint4*>(&Vt[r][half * 8]);
      vb_.q[1] = *reinterpret_cast<const uint4*>(&Vt[r][16 + half * 8]);
      o[ni] = __builtin_amdgcn_wmma_f32_16x16x32_bf16(
          false, pa.v, false, vb_.v, (short)0, o[ni], false, false);
    }
  }

  // normalize + store (acc rows = 8*half + j)
  #pragma unroll
  for (int ni = 0; ni < 4; ni++) {
    int col = h * HD + ni * 16 + l16;
    #pragma unroll
    for (int j = 0; j < 8; j++) {
      int row = blockIdx.x * 128 + wave * 16 + 8 * half + j;
      if (row < Nq) {
        float v = o[ni][j] / lrow[j];
        Ob[((size_t)(b * Nq + row)) * C_DIM + col] = f2bf(v);
      }
    }
  }
}

// ---------------------------------------------------------------------------
// Stage-1 "clip" attention: attends over the batch axis (sequence length 8).
// One wave per (s, h); lane = (b_q = lane>>2, 16-dim chunk = lane&3).
// qkv: (8, 196, 2304) bf16, o: (8, 196, 768) bf16.
// ---------------------------------------------------------------------------
__global__ void k_attn_b8(const unsigned short* __restrict__ qkv,
                          unsigned short* __restrict__ o) {
  int wid = blockIdx.x * 8 + (threadIdx.x >> 5);  // 0..2351 exactly
  int lane = threadIdx.x & 31;
  int s = wid / NH, h = wid % NH;
  int bq = lane >> 2;
  int d0 = (lane & 3) * 16;

  float q[16];
  const unsigned short* qp = qkv + ((size_t)(bq * 196 + s)) * 2304 + h * 64 + d0;
  #pragma unroll
  for (int i = 0; i < 16; i++) q[i] = bf2f(qp[i]);

  float sc[8];
  #pragma unroll
  for (int bk = 0; bk < 8; bk++) {
    const unsigned short* kp = qkv + ((size_t)(bk * 196 + s)) * 2304 + 768 + h * 64 + d0;
    float p = 0.f;
    #pragma unroll
    for (int i = 0; i < 16; i++) p += q[i] * bf2f(kp[i]);
    p += __shfl_xor(p, 1, 32);
    p += __shfl_xor(p, 2, 32);
    sc[bk] = p * ATT_SCALE;
  }
  float mx = sc[0];
  #pragma unroll
  for (int i = 1; i < 8; i++) mx = fmaxf(mx, sc[i]);
  float sum = 0.f;
  #pragma unroll
  for (int i = 0; i < 8; i++) { sc[i] = __expf(sc[i] - mx); sum += sc[i]; }
  float inv = 1.0f / sum;

  float acc[16];
  #pragma unroll
  for (int i = 0; i < 16; i++) acc[i] = 0.f;
  #pragma unroll
  for (int bk = 0; bk < 8; bk++) {
    const unsigned short* vp = qkv + ((size_t)(bk * 196 + s)) * 2304 + 1536 + h * 64 + d0;
    float w = sc[bk] * inv;
    #pragma unroll
    for (int i = 0; i < 16; i++) acc[i] += w * bf2f(vp[i]);
  }
  unsigned short* op = o + ((size_t)(bq * 196 + s)) * 768 + h * 64 + d0;
  #pragma unroll
  for (int i = 0; i < 16; i++) op[i] = f2bf(acc[i]);
}

// ---------------------------------------------------------------------------
// Host orchestration
// ---------------------------------------------------------------------------
static inline void launch_cvt(hipStream_t st, const float* x, unsigned short* y, int n) {
  int blocks = (n + 1023) / 1024;
  k_f32_to_bf16<<<blocks, 256, 0, st>>>(x, y, n);
}
static inline void launch_copy(hipStream_t st, const float* x, float* y, int n) {
  int blocks = (n + 1023) / 1024;
  k_copy_f32<<<blocks, 256, 0, st>>>(x, y, n);
}

extern "C" void kernel_launch(void* const* d_in, const int* in_sizes, int n_in,
                              void* d_out, int out_size, void* d_ws, size_t ws_size,
                              hipStream_t stream) {
  (void)in_sizes; (void)n_in; (void)out_size; (void)ws_size;

  const int S_TOK = 8 * 196;    // 1568
  const int T_TOK = 8 * 1568;   // 12544
  const int S_ELEM = S_TOK * C_DIM;
  const int T_ELEM = T_TOK * C_DIM;

  float* out_s = (float*)d_out;
  float* out_t = out_s + S_ELEM;

  // ---- workspace carve-up (deterministic) ----
  char* base = (char*)d_ws;
  size_t off = 0;
  auto alloc = [&](size_t bytes) -> char* {
    char* p = base + off;
    off = (off + bytes + 255) & ~(size_t)255;
    return p;
  };

  // weight order: {input idx, element count}
  const int widx[14]  = {4, 6, 10, 13, 17, 19, 21, 25, 27, 29, 33, 35, 39, 41};
  const int welem[14] = {2304*768, 768*768, 2304*768, 768*768, 768*768, 1536*768,
                         768*768, 768*768, 1536*768, 768*768,
                         3072*768, 768*3072, 3072*768, 768*3072};
  unsigned short* wbf[14];
  for (int i = 0; i < 14; i++) wbf[i] = (unsigned short*)alloc((size_t)welem[i] * 2);

  unsigned short* LNB  = (unsigned short*)alloc((size_t)T_ELEM * 2);
  unsigned short* QBUF = (unsigned short*)alloc((size_t)T_ELEM * 2);
  unsigned short* OBUF = (unsigned short*)alloc((size_t)T_ELEM * 2);
  unsigned short* XBF  = (unsigned short*)alloc((size_t)T_ELEM * 2);
  unsigned short* BIG  = (unsigned short*)alloc((size_t)T_TOK * 3072 * 2);
  float* BIASA = (float*)alloc(2304 * 4);
  float* BIAST = (float*)alloc(1536 * 4);
  float* BIASS = (float*)alloc(1536 * 4);

  // ---- residual streams into d_out ----
  launch_copy(stream, (const float*)d_in[0], out_s, S_ELEM);
  launch_copy(stream, (const float*)d_in[1], out_t, T_ELEM);

  // ---- weights -> bf16 ----
  for (int i = 0; i < 14; i++)
    launch_cvt(stream, (const float*)d_in[widx[i]], wbf[i], welem[i]);

  // ---- composite biases ----
  k_bias3<<<9, 256, 0, stream>>>(BIASA, (const float*)d_in[11], 768,
                                 nullptr, 768, (const float*)d_in[12], 768);
  k_bias3<<<6, 256, 0, stream>>>(BIAST, nullptr, 768,
                                 (const float*)d_in[20], 768, nullptr, 0);
  k_bias3<<<6, 256, 0, stream>>>(BIASS, nullptr, 768,
                                 (const float*)d_in[28], 768, nullptr, 0);

  auto gemm = [&](int epi, const unsigned short* A, const unsigned short* W,
                  const float* bias, unsigned short* Ybf, float* Yres,
                  int M, int N, int K) {
    dim3 g(N / 64, (M + 127) / 128);
    switch (epi) {
      case EPI_BF16:  k_gemm_wmma<EPI_BF16 ><<<g, 256, 0, stream>>>(A, W, bias, Ybf, Yres, M, N, K); break;
      case EPI_RES:   k_gemm_wmma<EPI_RES  ><<<g, 256, 0, stream>>>(A, W, bias, Ybf, Yres, M, N, K); break;
      case EPI_QGELU: k_gemm_wmma<EPI_QGELU><<<g, 256, 0, stream>>>(A, W, bias, Ybf, Yres, M, N, K); break;
      default:        k_gemm_wmma<EPI_GELU ><<<g, 256, 0, stream>>>(A, W, bias, Ybf, Yres, M, N, K); break;
    }
  };
  auto flash = [&](const unsigned short* Q, int CQ, int qo,
                   const unsigned short* K, int CK, int ko,
                   const unsigned short* V, int CV, int vo,
                   unsigned short* O, int Nq, int Nk) {
    dim3 g((Nq + 127) / 128, NH, 8);
    k_flash_wmma<<<g, 256, 0, stream>>>(Q, CQ, qo, K, CK, ko, V, CV, vo, O, Nq, Nk);
  };

  // ---- Stage 1: clip self-attention (attends over batch axis, len 8) ----
  k_layernorm_bf16<<<S_TOK, 256, 0, stream>>>(out_s, (const float*)d_in[2], (const float*)d_in[3], LNB, S_TOK);
  gemm(EPI_BF16, LNB, wbf[0], (const float*)d_in[5], BIG, nullptr, S_TOK, 2304, 768);
  k_attn_b8<<<294, 256, 0, stream>>>(BIG, OBUF);
  gemm(EPI_RES, OBUF, wbf[1], (const float*)d_in[7], nullptr, out_s, S_TOK, 768, 768);

  // ---- Stage 2: video self-attention ----
  k_layernorm_bf16<<<T_TOK, 256, 0, stream>>>(out_t, (const float*)d_in[8], (const float*)d_in[9], LNB, T_TOK);
  gemm(EPI_BF16, LNB, wbf[2], BIASA, BIG, nullptr, T_TOK, 2304, 768);
  flash(BIG, 2304, 0, BIG, 2304, 768, BIG, 2304, 1536, OBUF, 1568, 1568);
  gemm(EPI_RES, OBUF, wbf[3], (const float*)d_in[14], nullptr, out_t, T_TOK, 768, 768);

  // ---- Stage 3: T2S cross (q from LN(s_x), kv from updated t_x) ----
  k_layernorm_bf16<<<S_TOK, 256, 0, stream>>>(out_s, (const float*)d_in[15], (const float*)d_in[16], LNB, S_TOK);
  gemm(EPI_BF16, LNB, wbf[4], (const float*)d_in[18], QBUF, nullptr, S_TOK, 768, 768);
  launch_cvt(stream, out_t, XBF, T_ELEM);
  gemm(EPI_BF16, XBF, wbf[5], BIAST, BIG, nullptr, T_TOK, 1536, 768);
  flash(QBUF, 768, 0, BIG, 1536, 0, BIG, 1536, 768, OBUF, 196, 1568);
  gemm(EPI_RES, OBUF, wbf[6], (const float*)d_in[22], nullptr, out_s, S_TOK, 768, 768);

  // ---- Stage 4: S2T cross (q from LN(t_x), kv from updated s_x) ----
  k_layernorm_bf16<<<T_TOK, 256, 0, stream>>>(out_t, (const float*)d_in[23], (const float*)d_in[24], LNB, T_TOK);
  gemm(EPI_BF16, LNB, wbf[7], (const float*)d_in[26], QBUF, nullptr, T_TOK, 768, 768);
  launch_cvt(stream, out_s, XBF, S_ELEM);
  gemm(EPI_BF16, XBF, wbf[8], BIASS, BIG, nullptr, S_TOK, 1536, 768);
  flash(QBUF, 768, 0, BIG, 1536, 0, BIG, 1536, 768, OBUF, 1568, 196);
  gemm(EPI_RES, OBUF, wbf[9], (const float*)d_in[30], nullptr, out_t, T_TOK, 768, 768);

  // ---- Stage 5: clip MLP (QuickGELU) ----
  k_layernorm_bf16<<<S_TOK, 256, 0, stream>>>(out_s, (const float*)d_in[31], (const float*)d_in[32], LNB, S_TOK);
  gemm(EPI_QGELU, LNB, wbf[10], (const float*)d_in[34], BIG, nullptr, S_TOK, 3072, 768);
  gemm(EPI_RES, BIG, wbf[11], (const float*)d_in[36], nullptr, out_s, S_TOK, 768, 3072);

  // ---- Stage 6: video MLP (exact GELU) ----
  k_layernorm_bf16<<<T_TOK, 256, 0, stream>>>(out_t, (const float*)d_in[37], (const float*)d_in[38], LNB, T_TOK);
  gemm(EPI_GELU, LNB, wbf[12], (const float*)d_in[40], BIG, nullptr, T_TOK, 3072, 768);
  gemm(EPI_RES, BIG, wbf[13], (const float*)d_in[42], nullptr, out_t, T_TOK, 768, 3072);
}
